// GRUQNetwork_3985729651296
// MI455X (gfx1250) — matrix operand order repaired and checked by reading, not compiled, not run on hardware
//
#include <hip/hip_runtime.h>
#include <math.h>

typedef __bf16 bf16_t;
typedef __attribute__((ext_vector_type(16))) __bf16 v16bf;
typedef __attribute__((ext_vector_type(8)))  __bf16 v8bf;
typedef __attribute__((ext_vector_type(8)))  float  v8f;

#define T_DIM   32768
#define OBS_DIM 128
#define M_DIM   1024
#define R_DIM   1024
#define A_DIM   16
#define G3R     (3 * R_DIM)
#define SCAN_BLOCKS  12
#define SCAN_THREADS 256

__device__ __forceinline__ float bfu2f(unsigned short u) {
  unsigned v = ((unsigned)u) << 16;
  return __builtin_bit_cast(float, v);
}
__device__ __forceinline__ unsigned short f2bfu(float f) {
  unsigned u = __builtin_bit_cast(unsigned, f);
  u += 0x7FFFu + ((u >> 16) & 1u);   // round-to-nearest-even
  return (unsigned short)(u >> 16);
}
__device__ __forceinline__ float blo(unsigned d) { return __builtin_bit_cast(float, d << 16); }
__device__ __forceinline__ float bhi(unsigned d) { return __builtin_bit_cast(float, d & 0xFFFF0000u); }

__device__ __forceinline__ float mishf(float x) {
  float sp = (x > 20.0f) ? x : log1pf(expf(x));   // softplus with overflow guard
  return x * tanhf(sp);
}
__device__ __forceinline__ float sigmf(float x) { return 1.0f / (1.0f + expf(-x)); }

// ---------------- fp32 -> bf16 conversion ----------------
__global__ void f32_to_bf16_kernel(const float* __restrict__ in,
                                   unsigned short* __restrict__ out, long n) {
  long i = (long)blockIdx.x * blockDim.x + threadIdx.x;
  long stride = (long)gridDim.x * blockDim.x;
  for (; i < n; i += stride) out[i] = f2bfu(in[i]);
}

__global__ void init_bar_kernel(unsigned* bar) { *bar = 0u; }

// ---- fragment helpers (contiguous k order, same for A and B) ----
__device__ __forceinline__ v16bf load_a_frag(const bf16_t* Arow, int k0, int half) {
  // A 16x32 bf16: lanes 0-15 hold K 0..15, lanes 16-31 hold K 16..31 (contiguous)
  return *(const v16bf*)(Arow + k0 + half * 16);
}
__device__ __forceinline__ v16bf load_b_frag(const bf16_t* Brow, int k0, int half) {
  // B 32x16 bf16: lanes 0-15 hold K 0..15, lanes 16-31 hold K 16..31 (contiguous)
  return *(const v16bf*)(Brow + k0 + half * 16);
}

// ---------------- register-blocked WMMA GEMM: 64(M) x 32(N) macro-tile per wave ----------------
// C[M,N] = A[M,K] @ B[N,K]^T (+bias, +act).  8 WMMAs per 12 b128 loads per K-step.
__global__ __launch_bounds__(256)
void wmma_gemm_bf16_blk(const bf16_t* __restrict__ A, const bf16_t* __restrict__ B,
                        const float* __restrict__ bias,
                        float* __restrict__ Cf, unsigned short* __restrict__ Cb,
                        int Md, int Nd, int K, int act) {
  const int tilesN = Nd >> 5;                       // 32-wide in N
  const int tiles  = (Md >> 6) * tilesN;            // 64-tall in M
  const int tile   = blockIdx.x * 8 + (threadIdx.x >> 5);
  if (tile >= tiles) return;                        // wave-uniform
  const int lane = threadIdx.x & 31;
  const int half = lane >> 4;
  const int l16  = lane & 15;
  const int i0 = (tile / tilesN) << 6;
  const int j0 = (tile % tilesN) << 5;

  const bf16_t* Arow0 = A + (size_t)(i0 + l16) * K;
  const bf16_t* Arow1 = Arow0 + (size_t)16 * K;
  const bf16_t* Arow2 = Arow0 + (size_t)32 * K;
  const bf16_t* Arow3 = Arow0 + (size_t)48 * K;
  const bf16_t* Brow0 = B + (size_t)(j0 + l16) * K;
  const bf16_t* Brow1 = Brow0 + (size_t)16 * K;

  v8f acc[4][2] = {};
  for (int k0 = 0; k0 < K; k0 += 32) {
    v16bf a0 = load_a_frag(Arow0, k0, half);
    v16bf a1 = load_a_frag(Arow1, k0, half);
    v16bf a2 = load_a_frag(Arow2, k0, half);
    v16bf a3 = load_a_frag(Arow3, k0, half);
    v16bf b0 = load_b_frag(Brow0, k0, half);
    v16bf b1 = load_b_frag(Brow1, k0, half);
    if (k0 + 32 < K) {
      __builtin_prefetch((const void*)(Arow0 + k0 + 32), 0, 3);  // global_prefetch_b8
      __builtin_prefetch((const void*)(Brow0 + k0 + 32), 0, 3);
    }
    acc[0][0] = __builtin_amdgcn_wmma_f32_16x16x32_bf16(false, a0, false, b0, (short)0, acc[0][0], false, false);
    acc[0][1] = __builtin_amdgcn_wmma_f32_16x16x32_bf16(false, a0, false, b1, (short)0, acc[0][1], false, false);
    acc[1][0] = __builtin_amdgcn_wmma_f32_16x16x32_bf16(false, a1, false, b0, (short)0, acc[1][0], false, false);
    acc[1][1] = __builtin_amdgcn_wmma_f32_16x16x32_bf16(false, a1, false, b1, (short)0, acc[1][1], false, false);
    acc[2][0] = __builtin_amdgcn_wmma_f32_16x16x32_bf16(false, a2, false, b0, (short)0, acc[2][0], false, false);
    acc[2][1] = __builtin_amdgcn_wmma_f32_16x16x32_bf16(false, a2, false, b1, (short)0, acc[2][1], false, false);
    acc[3][0] = __builtin_amdgcn_wmma_f32_16x16x32_bf16(false, a3, false, b0, (short)0, acc[3][0], false, false);
    acc[3][1] = __builtin_amdgcn_wmma_f32_16x16x32_bf16(false, a3, false, b1, (short)0, acc[3][1], false, false);
  }

#pragma unroll
  for (int n = 0; n < 2; ++n) {
    const int col = j0 + n * 16 + l16;
    const float bcol = bias ? bias[col] : 0.0f;
#pragma unroll
    for (int m = 0; m < 4; ++m) {
#pragma unroll
      for (int r = 0; r < 8; ++r) {
        const int row = i0 + m * 16 + r + half * 8;
        float v = acc[m][n][r] + bcol;
        if (act == 1) v = mishf(v);
        const size_t idx = (size_t)row * Nd + col;
        if (Cf) Cf[idx] = v;
        if (Cb) Cb[idx] = f2bfu(v);
      }
    }
  }
}

// ---------------- simple WMMA GEMM (one 16x16 tile/wave) for narrow N (final layer) ----------------
__global__ __launch_bounds__(256)
void wmma_gemm_bf16(const bf16_t* __restrict__ A, const bf16_t* __restrict__ B,
                    const float* __restrict__ bias,
                    float* __restrict__ Cf, unsigned short* __restrict__ Cb,
                    int Md, int Nd, int K, int act) {
  const int tilesN = Nd >> 4;
  const int tiles  = (Md >> 4) * tilesN;
  const int tile   = blockIdx.x * 8 + (threadIdx.x >> 5);
  if (tile >= tiles) return;
  const int lane = threadIdx.x & 31;
  const int half = lane >> 4;
  const int l16  = lane & 15;
  const int i0 = (tile / tilesN) << 4;
  const int j0 = (tile % tilesN) << 4;

  const bf16_t* Arow = A + (size_t)(i0 + l16) * K;
  const bf16_t* Brow = B + (size_t)(j0 + l16) * K;

  v8f acc = {};
  for (int k0 = 0; k0 < K; k0 += 32) {
    v16bf av = load_a_frag(Arow, k0, half);
    v16bf bv = load_b_frag(Brow, k0, half);
    if (k0 + 32 < K) __builtin_prefetch((const void*)(Arow + k0 + 32), 0, 3);
    acc = __builtin_amdgcn_wmma_f32_16x16x32_bf16(false, av, false, bv, (short)0, acc, false, false);
  }

  const int col = j0 + l16;
  const float bcol = bias ? bias[col] : 0.0f;
#pragma unroll
  for (int r = 0; r < 8; ++r) {
    const int row = i0 + r + half * 8;
    float v = acc[r] + bcol;
    if (act == 1) v = mishf(v);
    const size_t idx = (size_t)row * Nd + col;
    if (Cf) Cf[idx] = v;
    if (Cb) Cb[idx] = f2bfu(v);
  }
}

// ---------------- sequential GRU scan: 12 persistent blocks, 1 global barrier/step ----------------
__global__ __launch_bounds__(SCAN_THREADS)
void gru_scan_kernel(const unsigned short* __restrict__ ig,   // [T,3R] bf16
                     const unsigned short* __restrict__ whh,  // [3R,R] bf16
                     const float* __restrict__ b_n,           // [R]
                     const float* __restrict__ s0,            // [R]
                     float* __restrict__ hg_base,             // [2,3R] double buffer
                     unsigned* __restrict__ bar,              // monotonic barrier counter
                     unsigned short* __restrict__ states,     // [T,R] bf16
                     float* __restrict__ final_state) {       // [R]
  __shared__ float s_sh[R_DIM];
  const int tid = threadIdx.x;
  for (int i = tid; i < R_DIM; i += SCAN_THREADS) s_sh[i] = s0[i];
  __syncthreads();

  const int grow = blockIdx.x * SCAN_THREADS + tid;           // 0..3071: one w_hh row/thread
  const uint4* wrow = (const uint4*)(whh + (size_t)grow * R_DIM);

  for (int t = 0; t < T_DIM; ++t) {
    float* hg = hg_base + (size_t)(t & 1) * G3R;
    float acc = 0.f;
#pragma unroll 4
    for (int j = 0; j < R_DIM / 8; ++j) {
      uint4 q = wrow[j];
      const float* sp = &s_sh[j * 8];
      acc = fmaf(blo(q.x), sp[0], acc); acc = fmaf(bhi(q.x), sp[1], acc);
      acc = fmaf(blo(q.y), sp[2], acc); acc = fmaf(bhi(q.y), sp[3], acc);
      acc = fmaf(blo(q.z), sp[4], acc); acc = fmaf(bhi(q.z), sp[5], acc);
      acc = fmaf(blo(q.w), sp[6], acc); acc = fmaf(bhi(q.w), sp[7], acc);
    }
    hg[grow] = acc;
    __threadfence();
    __syncthreads();
    if (tid == 0) {
      atomicAdd(bar, 1u);
      const unsigned target = (unsigned)(t + 1) * SCAN_BLOCKS;
      while (__hip_atomic_load(bar, __ATOMIC_ACQUIRE, __HIP_MEMORY_SCOPE_AGENT) < target)
        __builtin_amdgcn_s_sleep(8);
    }
    __syncthreads();
    __threadfence();
    const size_t igoff = (size_t)t * G3R;
#pragma unroll
    for (int rep = 0; rep < 4; ++rep) {
      const int i = tid + rep * SCAN_THREADS;
      const float igr = bfu2f(ig[igoff + i]);
      const float igz = bfu2f(ig[igoff + R_DIM + i]);
      const float ign = bfu2f(ig[igoff + 2 * R_DIM + i]);
      const float r = sigmf(igr + hg[i]);
      const float z = sigmf(igz + hg[R_DIM + i]);
      const float n = tanhf(ign + r * (hg[2 * R_DIM + i] + b_n[i]));
      const float snew = n + z * (s_sh[i] - n);
      s_sh[i] = snew;
      if (blockIdx.x == 0) states[(size_t)t * R_DIM + i] = f2bfu(snew);
    }
    __syncthreads();
  }
  if (blockIdx.x == 0)
    for (int i = tid; i < R_DIM; i += SCAN_THREADS) final_state[i] = s_sh[i];
}

// ---------------- host orchestration ----------------
extern "C" void kernel_launch(void* const* d_in, const int* in_sizes, int n_in,
                              void* d_out, int out_size, void* d_ws, size_t ws_size,
                              hipStream_t stream) {
  (void)in_sizes; (void)n_in; (void)out_size; (void)ws_size;
  const float* x       = (const float*)d_in[0];
  const float* state   = (const float*)d_in[1];
  const float* pre_w   = (const float*)d_in[4];
  const float* pre_b   = (const float*)d_in[5];
  const float* w_ih    = (const float*)d_in[6];
  const float* w_hh    = (const float*)d_in[7];
  const float* b_ih    = (const float*)d_in[8];
  const float* b_n     = (const float*)d_in[9];
  const float* post1_w = (const float*)d_in[10];
  const float* post1_b = (const float*)d_in[11];
  const float* post2_w = (const float*)d_in[12];
  const float* post2_b = (const float*)d_in[13];
  const float* final_w = (const float*)d_in[14];
  const float* final_b = (const float*)d_in[15];
  float* out = (float*)d_out;

  char* ws = (char*)d_ws;
  size_t off = 0;
  auto alloc = [&](size_t bytes) -> char* {
    char* p = ws + off;
    off += (bytes + 255) & ~(size_t)255;
    return p;
  };
  unsigned short* xb      = (unsigned short*)alloc((size_t)T_DIM * OBS_DIM * 2);
  unsigned short* pre_wb  = (unsigned short*)alloc((size_t)M_DIM * OBS_DIM * 2);
  unsigned short* w_ihb   = (unsigned short*)alloc((size_t)G3R * M_DIM * 2);
  unsigned short* w_hhb   = (unsigned short*)alloc((size_t)G3R * R_DIM * 2);
  unsigned short* post1b  = (unsigned short*)alloc((size_t)M_DIM * R_DIM * 2);
  unsigned short* post2b  = (unsigned short*)alloc((size_t)M_DIM * M_DIM * 2);
  unsigned short* finalb  = (unsigned short*)alloc((size_t)A_DIM * M_DIM * 2);
  unsigned short* h_bf    = (unsigned short*)alloc((size_t)T_DIM * M_DIM * 2);   // reused as y2
  unsigned short* ig_bf   = (unsigned short*)alloc((size_t)T_DIM * G3R * 2);
  unsigned short* st_bf   = (unsigned short*)alloc((size_t)T_DIM * R_DIM * 2);
  unsigned short* y1_bf   = (unsigned short*)alloc((size_t)T_DIM * M_DIM * 2);
  float*          hg_ws   = (float*)alloc((size_t)2 * G3R * 4);
  unsigned*       bar     = (unsigned*)alloc(256);

  const dim3 cb(256);
  const dim3 cg(2048);
  f32_to_bf16_kernel<<<cg, cb, 0, stream>>>(x,       xb,     (long)T_DIM * OBS_DIM);
  f32_to_bf16_kernel<<<cg, cb, 0, stream>>>(pre_w,   pre_wb, (long)M_DIM * OBS_DIM);
  f32_to_bf16_kernel<<<cg, cb, 0, stream>>>(w_ih,    w_ihb,  (long)G3R * M_DIM);
  f32_to_bf16_kernel<<<cg, cb, 0, stream>>>(w_hh,    w_hhb,  (long)G3R * R_DIM);
  f32_to_bf16_kernel<<<cg, cb, 0, stream>>>(post1_w, post1b, (long)M_DIM * R_DIM);
  f32_to_bf16_kernel<<<cg, cb, 0, stream>>>(post2_w, post2b, (long)M_DIM * M_DIM);
  f32_to_bf16_kernel<<<cg, cb, 0, stream>>>(final_w, finalb, (long)A_DIM * M_DIM);
  init_bar_kernel<<<1, 1, 0, stream>>>(bar);

  auto blk_blocks = [](int Md, int Nd) { return ((Md / 64) * (Nd / 32) + 7) / 8; };

  // pre MLP: h = mish(x @ pre_w.T + pre_b)
  wmma_gemm_bf16_blk<<<blk_blocks(T_DIM, M_DIM), 256, 0, stream>>>(
      (const bf16_t*)xb, (const bf16_t*)pre_wb, pre_b,
      (float*)nullptr, h_bf, T_DIM, M_DIM, OBS_DIM, 1);
  // igates = h @ w_ih.T + b_ih
  wmma_gemm_bf16_blk<<<blk_blocks(T_DIM, G3R), 256, 0, stream>>>(
      (const bf16_t*)h_bf, (const bf16_t*)w_ihb, b_ih,
      (float*)nullptr, ig_bf, T_DIM, G3R, M_DIM, 0);
  // sequential GRU scan
  gru_scan_kernel<<<SCAN_BLOCKS, SCAN_THREADS, 0, stream>>>(
      ig_bf, w_hhb, b_n, state, hg_ws, bar, st_bf, out + (size_t)T_DIM * A_DIM);
  // post MLP
  wmma_gemm_bf16_blk<<<blk_blocks(T_DIM, M_DIM), 256, 0, stream>>>(
      (const bf16_t*)st_bf, (const bf16_t*)post1b, post1_b,
      (float*)nullptr, y1_bf, T_DIM, M_DIM, R_DIM, 1);
  wmma_gemm_bf16_blk<<<blk_blocks(T_DIM, M_DIM), 256, 0, stream>>>(
      (const bf16_t*)y1_bf, (const bf16_t*)post2b, post2_b,
      (float*)nullptr, h_bf, T_DIM, M_DIM, M_DIM, 1);
  // final: y = y2 @ final_w.T + final_b  -> f32 straight into d_out
  wmma_gemm_bf16<<<(((T_DIM / 16) * (A_DIM / 16)) + 7) / 8, 256, 0, stream>>>(
      (const bf16_t*)h_bf, (const bf16_t*)finalb, final_b,
      out, (unsigned short*)nullptr, T_DIM, A_DIM, M_DIM, 0);
}
